// MOTNet_90847148245116
// MI455X (gfx1250) — matrix operand order repaired and checked by reading, not compiled, not run on hardware
//
#include <hip/hip_runtime.h>
#include <hip/hip_bf16.h>

typedef __attribute__((ext_vector_type(16))) _Float16 v16h;
typedef __attribute__((ext_vector_type(8)))  _Float16 v8h;
typedef __attribute__((ext_vector_type(8)))  float    v8f;

#define NSAMP 16384
#define MT 4        // tiles per sample
#define CCH 1024    // channels per tile
#define GD 64       // channels per group

// ---------------------------------------------------------------------------
// Weight conversion kernels
// ---------------------------------------------------------------------------
__global__ __launch_bounds__(256) void cvt_f32_to_f16(const float* __restrict__ s,
                                                      _Float16* __restrict__ d, int n) {
  int i = blockIdx.x * 256 + threadIdx.x;
  if (i < n) d[i] = (_Float16)s[i];
}

// Pad W3 [2,256] -> [16,256] f16 (zero rows 2..15), b3 [2] -> [16] f32
__global__ __launch_bounds__(256) void pack_w3(const float* __restrict__ w3,
                                               const float* __restrict__ b3,
                                               _Float16* __restrict__ w3h,
                                               float* __restrict__ b3p) {
  int i = blockIdx.x * 256 + threadIdx.x;
  if (i < 16 * 256) {
    int r = i >> 8, c = i & 255;
    w3h[i] = (r < 2) ? (_Float16)w3[r * 256 + c] : (_Float16)0.0f;
  }
  if (i < 16) b3p[i] = (i < 2) ? b3[i] : 0.0f;
}

// ---------------------------------------------------------------------------
// Feature builder: one block (256 threads) per sample.
// Output xh[s, 0:256]   = grouped gram (g*16 + m*4 + k)
//        xh[s, 256:512] = positional sin/cos embedding (fidx*16 + m*4 + k)
// ---------------------------------------------------------------------------
__global__ __launch_bounds__(256) void build_features(
    const float* __restrict__ cur_im, const float* __restrict__ ref_im,
    const float* __restrict__ cur_dt, const float* __restrict__ ref_dt,
    _Float16* __restrict__ xh) {
  __shared__ float f[MT * CCH];   // 16 KB feature tile
  __shared__ float boxes[16];     // 4 boxes x 4 coords
  const int s = blockIdx.x;
  const int t = threadIdx.x;

  const float4* c4 = (const float4*)(cur_im + (size_t)s * CCH);
  ((float4*)f)[t] = c4[t];                                   // 1024 floats
  const float4* r4 = (const float4*)(ref_im + (size_t)s * 3 * CCH);
#pragma unroll
  for (int i = 0; i < 3; ++i)
    ((float4*)(f + CCH))[t + 256 * i] = r4[t + 256 * i];     // 3072 floats

  if (t < 16)
    boxes[t] = (t < 4) ? cur_dt[(size_t)s * 4 + t]
                       : ref_dt[(size_t)s * 12 + (t - 4)];
  __syncthreads();

  // ---- grouped Gram: thread t -> (g, m, k), 64-long dot product ----
  {
    const int g = t >> 4, m = (t >> 2) & 3, k = t & 3;
    const float* am = f + m * CCH + g * GD;
    const float* ak = f + k * CCH + g * GD;
    float acc = 0.0f;
#pragma unroll 8
    for (int d = 0; d < GD; ++d) acc += am[d] * ak[d];
    xh[(size_t)s * 512 + t] = (_Float16)acc;
  }

  // ---- positional embedding: thread t -> (fidx, m, k) ----
  {
    const int fidx = t >> 4, m = (t >> 2) & 3, k = t & 3;
    const int p = fidx >> 2, q = fidx & 3;
    const float xm0 = boxes[m * 4 + 0], ym0 = boxes[m * 4 + 1];
    const float xm1 = boxes[m * 4 + 2], ym1 = boxes[m * 4 + 3];
    const float xk0 = boxes[k * 4 + 0], yk0 = boxes[k * 4 + 1];
    const float xk1 = boxes[k * 4 + 2], yk1 = boxes[k * 4 + 3];
    const float wm = xm1 - xm0, hm = ym1 - ym0;
    const float wk = xk1 - xk0, hk = yk1 - yk0;
    const float cxm = 0.5f * (xm0 + xm1), cym = 0.5f * (ym0 + ym1);
    const float cxk = 0.5f * (xk0 + xk1), cyk = 0.5f * (yk0 + yk1);
    float v;
    if (p == 0)      v = __logf(fabsf((cxm - cxk) / wm) + 0.001f);
    else if (p == 1) v = __logf(fabsf((cym - cyk) / hm) + 0.001f);
    else if (p == 2) v = __logf(wm / wk);
    else             v = __logf(hm / hk);
    const float dim = (q & 1) ? 31.6227766017f : 1.0f;  // 1000^(0.5*r)
    const float val = v * 100.0f / dim;
    const float e = (q < 2) ? __sinf(val) : __cosf(val);
    xh[(size_t)s * 512 + 256 + t] = (_Float16)e;
  }
}

// ---------------------------------------------------------------------------
// WMMA GEMM: Y = act(X[N,KK] * W[O,KK]^T + bias).
//
// Block (8 waves) computes a 128-row x (16*NT)-col output tile. The B strip
// (NT*16 rows of W) is staged in LDS ONCE per 256-wide K-chunk and shared by
// all 8 waves -> weight L2 traffic drops 8x, and B fragment reads become
// ds_load_b128 with immediate offsets (row stride KC+8 halves => consecutive
// lanes sit 4 banks apart: conflict-free). A stays as per-wave global b128.
//
// Fragment layouts per ISA 7.12.2:
//  A (16x32 f16): lane&15 = M row; lanes<16 hold K {0..7,16..23},
//                 lanes>=16 hold K {8..15,24..31}.
//  B (32x16 f16): lane&15 = N col (= W row, contiguous); lanes<16 -> K {0..7,16..23},
//                 lanes>=16 -> K {8..15,24..31}.
// ---------------------------------------------------------------------------
template <int NT, int KK>
__global__ __launch_bounds__(256) void gemm_wmma_f16(
    const _Float16* __restrict__ X, const _Float16* __restrict__ W,
    const float* __restrict__ bias, _Float16* __restrict__ Yh,
    float* __restrict__ Yf, int O, int relu, int outCols) {
  constexpr int ROWS = NT * 16;                 // B strip rows
  constexpr int KC   = (KK > 256) ? 256 : KK;   // K chunk staged in LDS
  constexpr int LDK  = KC + 8;                  // padded row stride (halves)
  __shared__ _Float16 bs[ROWS * LDK];           // <= 33.8 KB

  const int ngroups = O / (16 * NT);
  const int bm  = blockIdx.x / ngroups;         // 8-M-tile group
  const int tgB = blockIdx.x % ngroups;         // N group
  const int wv  = threadIdx.x >> 5;
  const int lane = threadIdx.x & 31;
  const int l15 = lane & 15;
  const int hi  = lane >> 4;                    // 0 or 1

  const int tm = bm * 8 + wv;
  const _Float16* __restrict__ xrow   = X + (size_t)(tm * 16 + l15) * KK;
  const _Float16* __restrict__ wstrip = W + (size_t)tgB * ROWS * KK;
  // Per-lane LDS base: column row l15, K-half select by hi
  const _Float16* brow = bs + l15 * LDK + hi * 8;

  v8f c[NT];
#pragma unroll
  for (int j = 0; j < NT; ++j) c[j] = (v8f){};

#pragma unroll
  for (int kc = 0; kc < KK; kc += KC) {
    // ---- cooperative stage: ROWS x KC strip of W into LDS ----
#pragma unroll
    for (int i = 0; i < ROWS * KC / 8 / 256; ++i) {
      const int v = threadIdx.x + 256 * i;
      const int row = v / (KC / 8);
      const int c8  = v % (KC / 8);
      *(v8h*)(bs + row * LDK + c8 * 8) =
          *(const v8h*)(wstrip + (size_t)row * KK + kc + c8 * 8);
    }
    __syncthreads();

    // ---- WMMA over this K chunk ----
#pragma unroll
    for (int k0 = 0; k0 < KC; k0 += 32) {
      const int kA = kc + k0 + hi * 8;

      if ((k0 & 63) == 0 && kc + k0 + 128 < KK)  // compile-time folded
        __builtin_prefetch(xrow + kc + k0 + 128, 0, 3);

      v8h alo = *(const v8h*)(xrow + kA);
      v8h ahi = *(const v8h*)(xrow + kA + 16);
      v16h a;
#pragma unroll
      for (int i = 0; i < 8; ++i) { a[i] = alo[i]; a[i + 8] = ahi[i]; }

#pragma unroll
      for (int j = 0; j < NT; ++j) {
        // ds_load_b128 pair, immediate offsets (j*16*LDK + k0 <= ~26K halves)
        v8h blo = *(const v8h*)(brow + j * 16 * LDK + k0);
        v8h bhi = *(const v8h*)(brow + j * 16 * LDK + k0 + 16);
        v16h b;
#pragma unroll
        for (int i = 0; i < 8; ++i) { b[i] = blo[i]; b[i + 8] = bhi[i]; }
        c[j] = __builtin_amdgcn_wmma_f32_16x16x32_f16(
            false, a, false, b, (short)0, c[j], false, false);
      }
    }
    __syncthreads();  // protect LDS before next chunk's staging
  }

  // D layout: VGPR r -> row = r + 8*hi, col = lane&15 within tile
#pragma unroll
  for (int j = 0; j < NT; ++j) {
    const int col = tgB * ROWS + j * 16 + l15;
    const float bv = bias[col];
#pragma unroll
    for (int r = 0; r < 8; ++r) {
      const int row = tm * 16 + r + hi * 8;
      float v = c[j][r] + bv;
      if (relu) v = v > 0.0f ? v : 0.0f;
      if (Yf) {
        if (col < outCols) Yf[(size_t)row * outCols + col] = v;
      } else {
        Yh[(size_t)row * O + col] = (_Float16)v;
      }
    }
  }
}

// ---------------------------------------------------------------------------
extern "C" void kernel_launch(void* const* d_in, const int* in_sizes, int n_in,
                              void* d_out, int out_size, void* d_ws, size_t ws_size,
                              hipStream_t stream) {
  const float* cur_im = (const float*)d_in[0];
  const float* ref_im = (const float*)d_in[1];
  const float* cur_dt = (const float*)d_in[2];
  const float* ref_dt = (const float*)d_in[3];
  const float* W1 = (const float*)d_in[4];
  const float* b1 = (const float*)d_in[5];
  const float* W2 = (const float*)d_in[6];
  const float* b2 = (const float*)d_in[7];
  const float* W3 = (const float*)d_in[8];
  const float* b3 = (const float*)d_in[9];

  char* p = (char*)d_ws;
  _Float16* xh  = (_Float16*)p;  p += (size_t)NSAMP * 512 * sizeof(_Float16);
  _Float16* h1h = (_Float16*)p;  p += (size_t)NSAMP * 512 * sizeof(_Float16);
  _Float16* h2h = (_Float16*)p;  p += (size_t)NSAMP * 256 * sizeof(_Float16);
  _Float16* w1h = (_Float16*)p;  p += (size_t)512 * 512 * sizeof(_Float16);
  _Float16* w2h = (_Float16*)p;  p += (size_t)256 * 512 * sizeof(_Float16);
  _Float16* w3h = (_Float16*)p;  p += (size_t)16 * 256 * sizeof(_Float16);
  float*    b3p = (float*)p;     p += 256;

  // Weight conversion (tiny; L2-resident thereafter)
  cvt_f32_to_f16<<<(512 * 512 + 255) / 256, 256, 0, stream>>>(W1, w1h, 512 * 512);
  cvt_f32_to_f16<<<(256 * 512 + 255) / 256, 256, 0, stream>>>(W2, w2h, 256 * 512);
  pack_w3<<<16, 256, 0, stream>>>(W3, b3, w3h, b3p);

  // Feature build: 268 MB HBM read -> 16 MB f16 activations (HBM-bound pass)
  build_features<<<NSAMP, 256, 0, stream>>>(cur_im, ref_im, cur_dt, ref_dt, xh);

  // Layer 1: [16384,512] x [512,512]^T, relu. 128 M-groups x 8 N-groups
  gemm_wmma_f16<4, 512><<<128 * 8, 256, 0, stream>>>(xh, w1h, b1, h1h, nullptr,
                                                     512, 1, 0);
  // Layer 2: [16384,512] x [256,512]^T, relu. 128 x 4
  gemm_wmma_f16<4, 512><<<128 * 4, 256, 0, stream>>>(h1h, w2h, b2, h2h, nullptr,
                                                     256, 1, 0);
  // Layer 3: [16384,256] x [16,256]^T (padded), cols 0..1 -> d_out fp32. 128 x 1
  gemm_wmma_f16<1, 256><<<128 * 1, 256, 0, stream>>>(h2h, w3h, b3p, nullptr,
                                                     (float*)d_out, 16, 0, 2);
}
